// GATLayer_55052890800667
// MI455X (gfx1250) — matrix operand order repaired and checked by reading, not compiled, not run on hardware
//
#include <hip/hip_runtime.h>

// GAT layer: B=8, N=1024, IN=256, OUT=256, H=4, D=64
// d_in order: h[8,1024,256] f32, adj[8,1024,1024] i32, W[256,256] f32,
//             a_src[4,64] f32, a_dst[4,64] f32, gamma[256] f32, beta[256] f32
// d_out: [8,1024,256] f32

#define GB 8
#define GN 1024
#define GK 256     // IN_DIM
#define GO 256     // OUT_DIM
#define GH 4
#define GD 64
#define ROWS (GB*GN)   // 8192

typedef __attribute__((ext_vector_type(2)))  float        v2f;
typedef __attribute__((ext_vector_type(8)))  float        v8f;
typedef __attribute__((ext_vector_type(16))) _Float16     v16h;
typedef __attribute__((ext_vector_type(4)))  unsigned int u32x4;
typedef __attribute__((ext_vector_type(8)))  int          i32x8;
typedef __attribute__((ext_vector_type(4)))  int          i32x4;

// ---- TDM availability / arity detection -----------------------------------
#if defined(__has_builtin)
#  if __has_builtin(__builtin_amdgcn_tensor_load_to_lds) && \
      __has_builtin(__builtin_amdgcn_s_wait_tensorcnt)
#    define CDNA5_HAS_TDM 1
#  endif
#endif
#ifndef CDNA5_HAS_TDM
#  define CDNA5_HAS_TDM 0
#endif
// therock-10.0 headers (6-arg builtin) ship this header; ROCm 7.2 (5-arg) doesn't.
#if __has_include(<hip/amd_detail/amd_gfx1250_TDM.h>)
#  define CDNA5_TDM_6ARG 1
#else
#  define CDNA5_TDM_6ARG 0
#endif

// ---------------------------------------------------------------------------
// Kernel 1: Wh = h @ W^T  (fp32 WMMA 16x16x4, full precision for residual path)
// ---------------------------------------------------------------------------
__global__ void __launch_bounds__(256)
wh_gemm_kernel(const float* __restrict__ h, const float* __restrict__ W,
               float* __restrict__ WhF, _Float16* __restrict__ WhH) {
  const int mtile = blockIdx.x;
  const int wave  = threadIdx.x >> 5;
  const int lane  = threadIdx.x & 31;
  const int m0    = mtile * 16;
  const int rowA  = lane & 15;          // A-matrix M row / B-matrix N col
  const int half  = lane >> 4;
  const int kgrp  = half * 2;           // A: VGPR0 = K0(lo half)/K2(hi half)

  const float* __restrict__ hrow = h + (size_t)(m0 + rowA) * GK;

  for (int nt = 0; nt < 2; ++nt) {
    const int n0 = (wave * 2 + nt) * 16;
    const float* __restrict__ wrow = W + (size_t)(n0 + rowA) * GK; // B[k][n]=W[n][k]
    v8f acc = {};
#pragma unroll 4
    for (int k0 = 0; k0 < GK; k0 += 4) {
      v2f a, b;
      a.x = hrow[k0 + kgrp];
      a.y = hrow[k0 + kgrp + 1];
      b.x = wrow[k0 + kgrp];
      b.y = wrow[k0 + kgrp + 1];
      acc = __builtin_amdgcn_wmma_f32_16x16x4_f32(
              false, a, false, b, (short)0, acc, false, false);
    }
    const int n     = n0 + rowA;
    const int mbase = m0 + half * 8;     // C/D: VGPR r -> (M=r+8*half, N=lane&15)
#pragma unroll
    for (int r = 0; r < 8; ++r) {
      const size_t idx = (size_t)(mbase + r) * GO + n;
      WhF[idx] = acc[r];
      WhH[idx] = (_Float16)acc[r];
    }
  }
}

// ---------------------------------------------------------------------------
// Kernel 1.5: repack WhH into the per-lane B-matrix register image.
// WhB record layout: [b][h][jblk(32)][dt(4)][lane(32)] -> 16 f16 (32 bytes),
// exactly the v16h a wave lane needs for the 32x16 f16 B operand.
// ---------------------------------------------------------------------------
__global__ void __launch_bounds__(256)
whb_repack_kernel(const _Float16* __restrict__ WhH, _Float16* __restrict__ WhB) {
  const int rec = blockIdx.x * blockDim.x + threadIdx.x;  // 8*4*32*4*32 = 131072
  if (rec >= GB * GH * 32 * 4 * 32) return;
  const int lane = rec & 31;
  const int dt   = (rec >> 5) & 3;
  const int jblk = (rec >> 7) & 31;
  const int hh   = (rec >> 12) & 3;
  const int b    = rec >> 14;
  const int n    = lane & 15;
  const int half = lane >> 4;
  const int j0   = jblk * 32 + 8 * half;

  _Float16* __restrict__ dst = WhB + (size_t)rec * 16;
  const _Float16* __restrict__ src =
      WhH + (size_t)b * GN * GO + hh * GD + dt * 16 + n;
#pragma unroll
  for (int r = 0; r < 8; ++r) {
    dst[r]     = src[(size_t)(j0 + r)      * GO];  // K = j0+r
    dst[r + 8] = src[(size_t)(j0 + 16 + r) * GO];  // K = j0+16+r
  }
}

// ---------------------------------------------------------------------------
// Kernel 2: e_src[b,h,n] = Wh[b,n,h,:].a_src[h], e_dst likewise.
// ---------------------------------------------------------------------------
__global__ void __launch_bounds__(256)
attn_coef_kernel(const float* __restrict__ WhF,
                 const float* __restrict__ a_src, const float* __restrict__ a_dst,
                 float* __restrict__ e_src, float* __restrict__ e_dst) {
  const int idx = blockIdx.x * blockDim.x + threadIdx.x;   // b*4096 + n*4 + h
  if (idx >= GB * GN * GH) return;
  const int hh = idx & 3;
  const int n  = (idx >> 2) & (GN - 1);
  const int b  = idx >> 12;
  const float* __restrict__ w = WhF + ((size_t)(b * GN + n)) * GO + hh * GD;
  float s = 0.f, t = 0.f;
#pragma unroll 8
  for (int d = 0; d < GD; ++d) {
    const float v = w[d];
    s += v * a_src[hh * GD + d];
    t += v * a_dst[hh * GD + d];
  }
  const int o = (b * GH + hh) * GN + n;
  e_src[o] = s;
  e_dst[o] = t;
}

// ---------------------------------------------------------------------------
// Kernel 3: per-(b,h,i) masked-softmax stats (one wave32 per row).
// ---------------------------------------------------------------------------
__global__ void __launch_bounds__(256)
row_stats_kernel(const int* __restrict__ adj,
                 const float* __restrict__ e_src, const float* __restrict__ e_dst,
                 float* __restrict__ mrow, float* __restrict__ linv) {
  const int gwave = (blockIdx.x * blockDim.x + threadIdx.x) >> 5;
  const int lane  = threadIdx.x & 31;
  const int i  = gwave & (GN - 1);
  const int hh = (gwave >> 10) & 3;
  const int b  = gwave >> 12;

  const float ei = e_src[gwave];
  const float* __restrict__ s    = e_dst + (size_t)(b * GH + hh) * GN;
  const int*   __restrict__ arow = adj + ((size_t)b * GN + i) * GN;

  const float NEG_INF = -__builtin_inff();
  float mx = NEG_INF, sum = 0.f;
  for (int j = lane; j < GN; j += 32) {
    if (arow[j] != 0) {
      float v = ei + s[j];
      v = v > 0.f ? v : 0.2f * v;            // leaky_relu(0.2)
      if (v > mx) { sum *= __expf(mx - v); mx = v; }
      sum += __expf(v - mx);
    }
  }
#pragma unroll
  for (int off = 16; off; off >>= 1) {
    const float mo = __shfl_xor(mx, off, 32);
    const float so = __shfl_xor(sum, off, 32);
    const float nm = fmaxf(mx, mo);
    float ns = 0.f;
    if (nm > NEG_INF) ns = sum * __expf(mx - nm) + so * __expf(mo - nm);
    mx = nm; sum = ns;
  }
  if (lane == 0) {
    mrow[gwave] = mx;
    linv[gwave] = (sum > 0.f) ? 1.f / sum : 0.f;
  }
}

// ---------------------------------------------------------------------------
// Kernel 4: out_pre[b,i,h,:] = softmax-row . Wh  (f16 WMMA 16x16x32, f32 acc)
// grid (N/16, B); 128 threads = 4 waves = 1 wave per head.
// LDS: adj tile byte-packed (16 KB, manual) + e_dst table (16 KB, via TDM DMA).
// B operand comes pre-swizzled from WhB: one aligned 32B v16h load per tile.
// ---------------------------------------------------------------------------
__global__ void __launch_bounds__(128)
attn_aggregate_kernel(const int* __restrict__ adj, const v16h* __restrict__ WhB,
                      const float* __restrict__ e_src, const float* __restrict__ e_dst,
                      const float* __restrict__ mrow, const float* __restrict__ linv,
                      float* __restrict__ out_pre) {
  __shared__ unsigned char adjT[16 * GN];   // 16 KB
  __shared__ float sdst[GH * GN];           // 16 KB, filled by TDM

  const int b   = blockIdx.y;
  const int i0  = blockIdx.x * 16;
  const int tid = threadIdx.x;

#if CDNA5_HAS_TDM
  // TDM: async DMA e_dst[b, :, :] (4096 f32 = 16 KB) global -> LDS.
  if (tid == 0) {
    const unsigned long long ga =
        (unsigned long long)(size_t)(e_dst + (size_t)b * GH * GN);
    // ISA aperture mapping: flat LDS address low 32 bits == LDS byte offset.
    const unsigned lds = (unsigned)(size_t)(&sdst[0]);
    u32x4 g0;
    g0.x = 1u;                                           // count=1, user mode
    g0.y = lds;                                          // lds_addr
    g0.z = (unsigned)(ga & 0xffffffffull);               // global_addr[31:0]
    g0.w = (unsigned)((ga >> 32) & 0x1ffffffull)         // global_addr[56:32]
         | (2u << 30);                                   // type=2 ("image")
    i32x8 g1;
    g1[0] = (2 << 16);            // workgroup_mask=0, data_size=2 (4 bytes)
    g1[1] = (int)(4096u << 16);   // tensor_dim0 = 4096 (lo16 @ bits[31:16])
    g1[2] = (int)(1u << 16);      // tensor_dim0 hi=0 | tensor_dim1 = 1
    g1[3] = (int)(4096u << 16);   // tensor_dim1 hi=0 | tile_dim0 = 4096
    g1[4] = 1;                    // tile_dim1 = 1, tile_dim2 = 0
    g1[5] = 4096;                 // tensor_dim0_stride[31:0]
    g1[6] = 0;                    // stride hi | tensor_dim1_stride lo
    g1[7] = 0;
    const i32x4 z4 = {0, 0, 0, 0};
#if CDNA5_TDM_6ARG
    const i32x8 z8 = {0, 0, 0, 0, 0, 0, 0, 0};
    __builtin_amdgcn_tensor_load_to_lds(g0, g1, z4, z4, z8, 0);
#else
    __builtin_amdgcn_tensor_load_to_lds(g0, g1, z4, z4, 0);
#endif
  }
  __builtin_amdgcn_s_wait_tensorcnt(0);
#else
  for (int idx = tid; idx < GH * GN; idx += 128)
    sdst[idx] = e_dst[(size_t)b * GH * GN + idx];
#endif

  // adj tile, byte-packed (coalesced int32 reads, L2-resident)
  const int* __restrict__ abase = adj + ((size_t)b * GN + i0) * GN;
  for (int idx = tid; idx < 16 * GN; idx += 128)
    adjT[idx] = (unsigned char)(abase[idx] != 0);
  __syncthreads();

  const int hh   = tid >> 5;    // wave == head
  const int lane = tid & 31;
  const int rowA = lane & 15;
  const int half = lane >> 4;
  const int i    = i0 + rowA;

  const float ei = e_src[(b * GH + hh) * GN + i];
  const float mi = mrow [(b * GH + hh) * GN + i];
  const float li = linv [(b * GH + hh) * GN + i];
  const unsigned char* __restrict__ arow = adjT + rowA * GN;
  const float* __restrict__ sj = sdst + hh * GN;
  // pre-swizzled B tiles for this (b, head): [jblk][dt][lane] v16h records
  const v16h* __restrict__ Bhead = WhB + (size_t)(b * GH + hh) * 32 * 4 * 32;

  v8f acc[4] = {v8f{}, v8f{}, v8f{}, v8f{}};

  for (int j0 = 0; j0 < GN; j0 += 32) {
    const int rbase = (j0 >> 5) * 4 * 32 + lane;
    if (j0 + 32 < GN)
      __builtin_prefetch((const void*)(Bhead + rbase + 4 * 32), 0, 0);

    // ---- A tile: probabilities p[16 x 32] in f16 (16-bit A layout) ----
    v16h a;
#pragma unroll
    for (int v = 0; v < 8; ++v) {
      const int koff = (v < 4) ? (2 * v + 8 * half) : (16 + 2 * (v - 4) + 8 * half);
#pragma unroll
      for (int t = 0; t < 2; ++t) {
        const int j = j0 + koff + t;
        float p = 0.f;
        if (arow[j]) {
          float x = ei + sj[j];
          x = x > 0.f ? x : 0.2f * x;
          p = __expf(x - mi) * li;
        }
        a[2 * v + t] = (_Float16)p;
      }
    }
    // ---- 4 d-subtiles: single v16h load per B tile, A reused ----
#pragma unroll
    for (int dt = 0; dt < 4; ++dt) {
      const v16h bm = Bhead[rbase + dt * 32];
      acc[dt] = __builtin_amdgcn_wmma_f32_16x16x32_f16(
                  false, a, false, bm, (short)0, acc[dt], false, false);
    }
  }

  const int n     = lane & 15;
  const int rb    = i0 + half * 8;   // C/D: VGPR r -> (M=r+8*half, N=lane&15)
#pragma unroll
  for (int dt = 0; dt < 4; ++dt)
#pragma unroll
    for (int r = 0; r < 8; ++r)
      out_pre[((size_t)(b * GN + rb + r)) * GO + hh * GD + dt * 16 + n] = acc[dt][r];
}

// ---------------------------------------------------------------------------
// Kernel 5/6: deterministic BatchNorm stats (no float atomics -> graph-stable)
// ---------------------------------------------------------------------------
#define BN_BLOCKS 64
__global__ void __launch_bounds__(256)
bn_partial_kernel(const float* __restrict__ out_pre,
                  float* __restrict__ psum, float* __restrict__ psumsq) {
  const int c  = threadIdx.x;
  const int r0 = blockIdx.x * (ROWS / BN_BLOCKS);
  float s = 0.f, q = 0.f;
  for (int r = 0; r < ROWS / BN_BLOCKS; ++r) {
    const float v = out_pre[(size_t)(r0 + r) * GO + c];
    s += v; q += v * v;
  }
  psum  [blockIdx.x * GO + c] = s;
  psumsq[blockIdx.x * GO + c] = q;
}

__global__ void __launch_bounds__(256)
bn_finalize_kernel(const float* __restrict__ psum, const float* __restrict__ psumsq,
                   float* __restrict__ meanv, float* __restrict__ invstd) {
  const int c = threadIdx.x;
  float s = 0.f, q = 0.f;
  for (int blk = 0; blk < BN_BLOCKS; ++blk) {
    s += psum[blk * GO + c];
    q += psumsq[blk * GO + c];
  }
  const float mean = s * (1.f / ROWS);
  const float var  = q * (1.f / ROWS) - mean * mean;   // biased var
  meanv [c] = mean;
  invstd[c] = rsqrtf(var + 1e-5f);
}

// ---------------------------------------------------------------------------
// Kernel 7: out = elu((x-mean)*invstd*gamma + beta) + Wh_flat
// ---------------------------------------------------------------------------
__global__ void __launch_bounds__(256)
bn_elu_residual_kernel(const float* __restrict__ out_pre, const float* __restrict__ WhF,
                       const float* __restrict__ meanv, const float* __restrict__ invstd,
                       const float* __restrict__ gamma, const float* __restrict__ beta,
                       float* __restrict__ out) {
  const size_t idx = (size_t)blockIdx.x * blockDim.x + threadIdx.x;
  if (idx >= (size_t)ROWS * GO) return;
  const int c = (int)(idx & (GO - 1));
  const float x = out_pre[idx];
  const float y = (x - meanv[c]) * invstd[c] * gamma[c] + beta[c];
  const float e = y > 0.f ? y : (__expf(y) - 1.f);
  out[idx] = e + WhF[idx];
}

// ---------------------------------------------------------------------------
extern "C" void kernel_launch(void* const* d_in, const int* in_sizes, int n_in,
                              void* d_out, int out_size, void* d_ws, size_t ws_size,
                              hipStream_t stream) {
  const float* h     = (const float*)d_in[0];
  const int*   adj   = (const int*)  d_in[1];
  const float* W     = (const float*)d_in[2];
  const float* a_src = (const float*)d_in[3];
  const float* a_dst = (const float*)d_in[4];
  const float* gamma = (const float*)d_in[5];
  const float* beta  = (const float*)d_in[6];
  float* out = (float*)d_out;

  // workspace layout (~25.5 MB total)
  char* w = (char*)d_ws;
  float*    WhF     = (float*)w;    w += (size_t)ROWS * GO * 4;      // 8 MiB
  _Float16* WhH     = (_Float16*)w; w += (size_t)ROWS * GO * 2;      // 4 MiB
  _Float16* WhB     = (_Float16*)w; w += (size_t)ROWS * GO * 2;      // 4 MiB (B-swizzled)
  float*    e_src   = (float*)w;    w += (size_t)GB * GH * GN * 4;   // 128 KiB
  float*    e_dst   = (float*)w;    w += (size_t)GB * GH * GN * 4;
  float*    mrow    = (float*)w;    w += (size_t)GB * GH * GN * 4;
  float*    linv    = (float*)w;    w += (size_t)GB * GH * GN * 4;
  float*    out_pre = (float*)w;    w += (size_t)ROWS * GO * 4;      // 8 MiB
  float*    psum    = (float*)w;    w += (size_t)BN_BLOCKS * GO * 4;
  float*    psumsq  = (float*)w;    w += (size_t)BN_BLOCKS * GO * 4;
  float*    meanv   = (float*)w;    w += (size_t)GO * 4;
  float*    invstd  = (float*)w;    w += (size_t)GO * 4;

  // 1. Wh = h @ W^T (fp32 WMMA) + f16 copy
  wh_gemm_kernel<<<ROWS / 16, 256, 0, stream>>>(h, W, WhF, WhH);
  // 1.5 repack f16 Wh into B-matrix lane records
  whb_repack_kernel<<<(GB * GH * 32 * 4 * 32) / 256, 256, 0, stream>>>(WhH, WhB);
  // 2. per-node head coefficients
  attn_coef_kernel<<<(GB * GN * GH + 255) / 256, 256, 0, stream>>>(WhF, a_src, a_dst, e_src, e_dst);
  // 3. masked softmax row stats (one wave per row, 32768 rows)
  row_stats_kernel<<<(GB * GH * GN * 32) / 256, 256, 0, stream>>>(adj, e_src, e_dst, mrow, linv);
  // 4. fused softmax-weights + aggregation (f16 WMMA + TDM staging)
  attn_aggregate_kernel<<<dim3(GN / 16, GB), 128, 0, stream>>>(
      adj, (const v16h*)WhB, e_src, e_dst, mrow, linv, out_pre);
  // 5-6. deterministic BatchNorm statistics
  bn_partial_kernel<<<BN_BLOCKS, 256, 0, stream>>>(out_pre, psum, psumsq);
  bn_finalize_kernel<<<1, 256, 0, stream>>>(psum, psumsq, meanv, invstd);
  // 7. BN apply + ELU + residual
  bn_elu_residual_kernel<<<(ROWS * GO) / 256, 256, 0, stream>>>(out_pre, WhF, meanv, invstd, gamma, beta, out);
}
